// VNGraphEncoderMultiScale_41120016892414
// MI455X (gfx1250) — matrix operand order repaired and checked by reading, not compiled, not run on hardware
//
#include <hip/hip_runtime.h>
#include <math.h>

typedef _Float16 v16h __attribute__((ext_vector_type(16)));
typedef _Float16 v8h  __attribute__((ext_vector_type(8)));
typedef float    v8f  __attribute__((ext_vector_type(8)));
typedef float    v4f  __attribute__((ext_vector_type(4)));

#define TPB 256

// ---------------------------------------------------------------------------
// Fragment helpers (this code's contiguous-K 16-bit order, wave32).
// A (16x32 f16): lane L row M=L%16, K = hi*16 + {0..15}
// B (32x16 f16): lane L col N=L%16, K = hi*16 + {0..15}   (hi = L>=16)
// ---------------------------------------------------------------------------
__device__ __forceinline__ v16h load_a16(const _Float16* __restrict__ A,
                                         size_t arow, int kb, int hi) {
  v8h lo = *(const v8h*)(A + arow + kb + hi * 16);
  v8h hh = *(const v8h*)(A + arow + kb + hi * 16 + 8);
  v16h a;
#pragma unroll
  for (int i = 0; i < 8; ++i) { a[i] = lo[i]; a[8 + i] = hh[i]; }
  return a;
}

__device__ __forceinline__ v16h cvt16(const float* __restrict__ p) {
  v16h b;
#pragma unroll
  for (int q = 0; q < 4; ++q) {
    v4f f = *(const v4f*)(p + q * 4);
#pragma unroll
    for (int i = 0; i < 4; ++i) b[q * 4 + i] = (_Float16)f[i];
  }
  return b;
}

// weight f32 -> f16 with zero padding to (Mp, Kp); Mp mult of 128, Kp of 32
__global__ void k_w2h(const float* __restrict__ W, _Float16* __restrict__ dst,
                      int M, int K, int Kp, long n) {
  long gid = (long)blockIdx.x * blockDim.x + threadIdx.x;
  if (gid >= n) return;
  int r = (int)(gid / Kp), c = (int)(gid % Kp);
  dst[gid] = (r < M && c < K) ? (_Float16)W[(size_t)r * K + c] : (_Float16)0.f;
}

// ---------------------------------------------------------------------------
// Generic WMMA GEMM: D(M,N) = A(M,K)*B(K,N). A: f16 padded (Mp,K), K%32==0.
// One wave: 1 column tile (16) x MT row tiles; B fragment loaded once,
// reused by MT WMMAs.
// BMODE 0: B[k][j] = Bsrc[j*ldB + k]      BMODE 1: Bsrc[(j/3)*K*3 + k*3 + j%3]
// DMODE 0: Dst[j*ldD + m] (+bias,+resid)  DMODE 1: Dst[(j/3)*Mstore*3+m*3+j%3]
// ---------------------------------------------------------------------------
template <int BMODE, int DMODE, int MT, bool BIAS, bool RESID>
__global__ void k_gemm(const _Float16* __restrict__ A, const float* __restrict__ Bsrc,
                       float* __restrict__ Dst, const float* __restrict__ bias,
                       int M, int N, int K, int ldB, int ldD, int Mstore) {
  int lane = threadIdx.x & 31;
  int wv   = threadIdx.x >> 5;
  int mt = (M + 15) >> 4;
  int nt = (N + 15) >> 4;
  int mg = (mt + MT - 1) / MT;
  long tile = (long)blockIdx.x * (blockDim.x >> 5) + wv;
  if (tile >= (long)mg * nt) return;
  int tg = (int)(tile % mg), tn = (int)(tile / mg);
  int tmBase = tg * MT;
  int hi   = (lane >= 16) ? 1 : 0;
  int hi16 = hi << 4;
  int ncol = tn * 16 + (lane & 15);
  bool jv  = (ncol < N);
  int ncl  = jv ? ncol : (N - 1);          // clamped for load addressing
  size_t arow0 = (size_t)(tmBase * 16 + (lane & 15)) * K;
  const size_t astep = (size_t)16 * K;
  const float* bptr;
  int e = 0, d = 0;
  if (BMODE == 0) {
    bptr = Bsrc + (size_t)ncl * ldB;
  } else {
    e = ncl / 3; d = ncl - e * 3;
    bptr = Bsrc + (size_t)e * (size_t)K * 3 + d;
  }
  v8f zero8 = {};
  v8f acc[MT];
#pragma unroll
  for (int u = 0; u < MT; ++u) acc[u] = zero8;
  int ksteps = K >> 5;
  for (int ks = 0; ks < ksteps; ++ks) {
    int kb = ks << 5;
    v16h b;
    if (BMODE == 0) {
      b = cvt16(bptr + kb + hi16);
      if (ks + 1 < ksteps) __builtin_prefetch(bptr + kb + 32 + hi16, 0, 0);
    } else {
      const float* bp = bptr + (size_t)(kb + hi16) * 3;
#pragma unroll
      for (int p = 0; p < 16; ++p) b[p] = (_Float16)bp[(size_t)p * 3];
    }
#pragma unroll
    for (int u = 0; u < MT; ++u) {
      v16h a = load_a16(A, arow0 + (size_t)u * astep, kb, hi);
      acc[u] = __builtin_amdgcn_wmma_f32_16x16x32_f16(false, a, false, b, (short)0,
                                                      acc[u], false, false);
    }
  }
#pragma unroll
  for (int u = 0; u < MT; ++u) {
    int m0 = (tmBase + u) * 16 + hi * 8;
    if (DMODE == 0) {
      if (jv && m0 < M) {
        if ((M & 15) == 0 || m0 + 7 < M) {
          float* dp = Dst + (size_t)ncol * ldD + m0;
          v4f o0, o1;
#pragma unroll
          for (int i = 0; i < 4; ++i) { o0[i] = acc[u][i]; o1[i] = acc[u][4 + i]; }
          if (BIAS)  { o0 += *(const v4f*)(bias + m0); o1 += *(const v4f*)(bias + m0 + 4); }
          if (RESID) { o0 += *(const v4f*)dp;          o1 += *(const v4f*)(dp + 4); }
          *(v4f*)dp = o0;
          *(v4f*)(dp + 4) = o1;
        } else {
#pragma unroll
          for (int r = 0; r < 8; ++r) {
            int m = m0 + r;
            if (m < M) {
              float vout = acc[u][r];
              if (BIAS) vout += bias[m];
              size_t di = (size_t)ncol * ldD + m;
              if (RESID) vout += Dst[di];
              Dst[di] = vout;
            }
          }
        }
      }
    } else {
#pragma unroll
      for (int r = 0; r < 8; ++r) {
        int m = m0 + r;
        if (m < Mstore && jv)
          Dst[(size_t)e * Mstore * 3 + (size_t)m * 3 + d] = acc[u][r];
      }
    }
  }
}

// ---------------------------------------------------------------------------
// vn_block #1 GEMM, fused edge gather. h layout (BN, 3, in_ch).
// Region boundaries (in_ch, 2*in_ch) are multiples of 32 when in_ch>=32, so
// every 32-wide K-step is wave-uniformly in one region; tail step is a pure
// select (no loads). in_ch==1 handled by an explicit 3-element path.
// MT covers all row tiles: B gathered exactly once per column.
// ---------------------------------------------------------------------------
template <int MT>
__global__ void k_vn1(const _Float16* __restrict__ A, const float* __restrict__ h,
                      const int* __restrict__ src, const int* __restrict__ tgt,
                      const float* __restrict__ pos, const float* __restrict__ dist,
                      float* __restrict__ Dst, int M, int N, int Kp, int in_ch) {
  int lane = threadIdx.x & 31;
  int wv   = threadIdx.x >> 5;
  int nt = (N + 15) >> 4;
  long tile = (long)blockIdx.x * (blockDim.x >> 5) + wv;
  if (tile >= (long)nt) return;
  int tn = (int)tile;
  int hi   = (lane >= 16) ? 1 : 0;
  int hi16 = hi << 4;
  int ncol = tn * 16 + (lane & 15);
  bool jv  = (ncol < N);
  int ncl  = jv ? ncol : (N - 1);
  size_t arow0 = (size_t)(lane & 15) * Kp;
  const size_t astep = (size_t)16 * Kp;
  int e = ncl / 3, d = ncl - e * 3;
  int si = src[e], ti = tgt[e];
  const float* hs = h + ((size_t)si * 3 + d) * in_ch;
  const float* ht = h + ((size_t)ti * 3 + d) * in_ch;
  float ru = (pos[ti * 3 + d] - pos[si * 3 + d]) / (dist[e] + 1e-8f);
  int in2 = 2 * in_ch;
  v8f zero8 = {};
  v8f acc[MT];
#pragma unroll
  for (int u = 0; u < MT; ++u) acc[u] = zero8;
  int ksteps = Kp >> 5;
  for (int ks = 0; ks < ksteps; ++ks) {
    int kb = ks << 5;
    int kq0 = kb + hi16;
    v16h b;
    if (kb + 31 < in_ch) {                       // uniform: all h_src
      b = cvt16(hs + kq0);
    } else if (kb >= in_ch && kb + 31 < in2) {   // uniform: all (h_tgt - h_src)
      int base = kq0 - in_ch;
#pragma unroll
      for (int q = 0; q < 4; ++q) {
        v4f ft = *(const v4f*)(ht + base + q * 4);
        v4f fs = *(const v4f*)(hs + base + q * 4);
#pragma unroll
        for (int i = 0; i < 4; ++i) b[q * 4 + i] = (_Float16)(ft[i] - fs[i]);
      }
    } else if (in_ch > 1) {                      // uniform tail: r_unit row only
#pragma unroll
      for (int p = 0; p < 16; ++p)
        b[p] = (_Float16)((kq0 + p == in2) ? ru : 0.f);
    } else {                                     // in_ch == 1 (K = 3, one step)
      float b0 = hs[0];
      float b1 = ht[0] - b0;
#pragma unroll
      for (int p = 0; p < 16; ++p) {
        int kq = kq0 + p;
        float bv = (kq == 0) ? b0 : (kq == 1) ? b1 : (kq == 2) ? ru : 0.f;
        b[p] = (_Float16)bv;
      }
    }
#pragma unroll
    for (int u = 0; u < MT; ++u) {
      v16h a = load_a16(A, arow0 + (size_t)u * astep, kb, hi);
      acc[u] = __builtin_amdgcn_wmma_f32_16x16x32_f16(false, a, false, b, (short)0,
                                                      acc[u], false, false);
    }
  }
  if (jv) {  // M % 16 == 0 always
#pragma unroll
    for (int u = 0; u < MT; ++u) {
      int m0 = u * 16 + hi * 8;
      float* dp = Dst + (size_t)ncol * M + m0;
      v4f o0, o1;
#pragma unroll
      for (int i = 0; i < 4; ++i) { o0[i] = acc[u][i]; o1[i] = acc[u][4 + i]; }
      *(v4f*)dp = o0;
      *(v4f*)(dp + 4) = o1;
    }
  }
}

// ---------------------------------------------------------------------------
// Elementwise / reduction kernels (x and h in (·, 3, C) layout)
// ---------------------------------------------------------------------------
__global__ void k_zero(float* p, long n) {
  long i = (long)blockIdx.x * blockDim.x + threadIdx.x;
  if (i < n) p[i] = 0.f;
}

__global__ void k_edge_geom(const int* __restrict__ src, const int* __restrict__ tgt,
                            const float* __restrict__ pos, float* __restrict__ dist,
                            float* __restrict__ sumd, float* __restrict__ cnt, int E) {
  int e = blockIdx.x * blockDim.x + threadIdx.x;
  if (e >= E) return;
  int s = src[e], t = tgt[e];
  float dx = pos[t * 3 + 0] - pos[s * 3 + 0];
  float dy = pos[t * 3 + 1] - pos[s * 3 + 1];
  float dz = pos[t * 3 + 2] - pos[s * 3 + 2];
  float di = sqrtf(dx * dx + dy * dy + dz * dz);
  dist[e] = di;
  atomicAdd(&sumd[s], di);
  atomicAdd(&cnt[s], 1.0f);
}

__global__ void k_node_ld(const float* __restrict__ sumd, const float* __restrict__ cnt,
                          float* __restrict__ ldv, int n) {
  int i = blockIdx.x * blockDim.x + threadIdx.x;
  if (i < n) ldv[i] = sumd[i] / fmaxf(cnt[i], 1.f);
}

// per-edge invariant MLP: 4 -> 16 (silu) -> 16 -> out_ch (sigmoid)
__global__ void k_mod(const int* __restrict__ src, const float* __restrict__ dist,
                      const float* __restrict__ ldv,
                      const float* __restrict__ W1, const float* __restrict__ b1,
                      const float* __restrict__ W2, const float* __restrict__ b2,
                      const float* __restrict__ mW, const float* __restrict__ mb,
                      float* __restrict__ mod, int E, int C) {
  __shared__ float s[352 + 128 * 17];
  float* sW1 = s;
  float* sb1 = s + 64;
  float* sW2 = s + 80;
  float* sb2 = s + 336;
  float* smW = s + 352;
  float* smb = s + 352 + C * 16;
  int tot = 352 + C * 17;
  for (int i = threadIdx.x; i < tot; i += blockDim.x) {
    float v;
    if (i < 64)        v = W1[i];
    else if (i < 80)   v = b1[i - 64];
    else if (i < 336)  v = W2[i - 80];
    else if (i < 352)  v = b2[i - 336];
    else if (i < 352 + C * 16) v = mW[i - 352];
    else               v = mb[i - 352 - C * 16];
    s[i] = v;
  }
  __syncthreads();
  int e = blockIdx.x * blockDim.x + threadIdx.x;
  if (e >= E) return;
  float di = dist[e];
  float l  = ldv[src[e]];
  float f0 = di, f1 = di / (l + 1e-6f), f2 = logf(di + 1e-6f), f3 = l;
  float h1[16], h2[16];
#pragma unroll
  for (int o = 0; o < 16; ++o) {
    float t = sb1[o] + sW1[o * 4] * f0 + sW1[o * 4 + 1] * f1 + sW1[o * 4 + 2] * f2 +
              sW1[o * 4 + 3] * f3;
    h1[o] = t / (1.f + expf(-t));
  }
#pragma unroll
  for (int o = 0; o < 16; ++o) {
    float t = sb2[o];
#pragma unroll
    for (int k = 0; k < 16; ++k) t += sW2[o * 16 + k] * h1[k];
    h2[o] = t;
  }
  float* mo = mod + (size_t)e * C;
  for (int o = 0; o < C; ++o) {
    float t = smb[o];
#pragma unroll
    for (int k = 0; k < 16; ++k) t += smW[o * 16 + k] * h2[k];
    mo[o] = 1.f / (1.f + expf(-t));
  }
}

// per-channel sum/sumsq of vector norms over edges; x layout (E,3,C)
__global__ void k_vn_stats(const float* __restrict__ x, float* __restrict__ stat,
                           int E, int C) {
  __shared__ float ls[256];
  int tid = threadIdx.x;
  for (int i = tid; i < 2 * C; i += blockDim.x) ls[i] = 0.f;
  __syncthreads();
  long gid = (long)blockIdx.x * blockDim.x + tid;
  if (gid < (long)E * C) {
    int e = (int)(gid / C), c = (int)(gid % C);
    const float* p = x + (size_t)e * 3 * C + c;
    float x0 = p[0], x1 = p[C], x2 = p[2 * C];
    float n = sqrtf(x0 * x0 + x1 * x1 + x2 * x2);
    atomicAdd(&ls[c], n);
    atomicAdd(&ls[C + c], n * n);
  }
  __syncthreads();
  for (int i = tid; i < 2 * C; i += blockDim.x)
    if (ls[i] != 0.f) atomicAdd(&stat[i], ls[i]);
}

// VN normalization + direction-gated leaky relu; x layout (E,3,C)
__global__ void k_vn_norm(float* __restrict__ x, const float* __restrict__ stat,
                          const float* __restrict__ gamma, const float* __restrict__ dir,
                          int E, int C) {
  long gid = (long)blockIdx.x * blockDim.x + threadIdx.x;
  if (gid >= (long)E * C) return;
  int e = (int)(gid / C), c = (int)(gid % C);
  float* p = x + (size_t)e * 3 * C + c;
  float x0 = p[0], x1 = p[C], x2 = p[2 * C];
  float nrm = sqrtf(x0 * x0 + x1 * x1 + x2 * x2);
  float fE = (float)E;
  float mean = stat[c] / fE;
  float var  = (stat[C + c] - fE * mean * mean) / fmaxf(fE - 1.f, 1.f);
  float std_ = sqrtf(fmaxf(var, 0.f)) + 1e-5f;
  float sc = (1.f / (nrm + 1e-5f)) * (nrm - mean) / std_ * gamma[c];
  x0 *= sc; x1 *= sc; x2 *= sc;
  float dx = dir[c * 3], dy = dir[c * 3 + 1], dz = dir[c * 3 + 2];
  float dn = fmaxf(sqrtf(dx * dx + dy * dy + dz * dz), 1e-12f);
  float proj = (x0 * dx + x1 * dy + x2 * dz) / dn;
  float sl = (proj >= 0.f) ? 1.f : 0.2f;
  p[0] = x0 * sl; p[C] = x1 * sl; p[2 * C] = x2 * sl;
}

// modulate + segment-sum into node features; h layout (BN,3,C)
__global__ void k_scatter(const float* __restrict__ x, const float* __restrict__ mod,
                          const int* __restrict__ src, float* __restrict__ h,
                          int E, int C) {
  long gid = (long)blockIdx.x * blockDim.x + threadIdx.x;
  if (gid >= (long)E * C) return;
  int e = (int)(gid / C), c = (int)(gid % C);
  float m = mod[(size_t)e * C + c];
  const float* p = x + (size_t)e * 3 * C + c;
  float* hb = h + (size_t)src[e] * 3 * C + c;
  atomicAdd(&hb[0],     p[0] * m);
  atomicAdd(&hb[C],     p[C] * m);
  atomicAdd(&hb[2 * C], p[2 * C] * m);
}

__global__ void k_divcnt(float* __restrict__ h, const float* __restrict__ cnt,
                         int BN, int C) {
  long gid = (long)blockIdx.x * blockDim.x + threadIdx.x;
  if (gid >= (long)BN * C) return;
  int n = (int)(gid / C), c = (int)(gid % C);
  float s = 1.f / fmaxf(cnt[n], 1.f);
  float* p = h + (size_t)n * 3 * C + c;
  p[0] *= s; p[C] *= s; p[2 * C] *= s;
}

// tokens[n][s*384 + c*3+d] = h[n][d][c]   (h: (BN,3,128))
__global__ void k_copy_tok(const float* __restrict__ h, float* __restrict__ tok,
                           int s, int BN) {
  long gid = (long)blockIdx.x * blockDim.x + threadIdx.x;
  if (gid >= (long)BN * 384) return;
  int n = (int)(gid / 384), q = (int)(gid % 384);
  int c = q / 3, d = q - c * 3;
  tok[(size_t)n * 1152 + s * 384 + q] = h[((size_t)n * 3 + d) * 128 + c];
}

// per (node, head, ti) attention over T=3 tokens, hd=96
__global__ void k_attn(const float* __restrict__ qkv, float* __restrict__ ao, int BN) {
  int gid = blockIdx.x * blockDim.x + threadIdx.x;
  if (gid >= BN * 12) return;
  int ti = gid % 3, nh = gid / 3;
  int head = nh % 4, n = nh / 4;
  const float* base = qkv + (size_t)n * 3 * 1152;
  const float* q  = base + ti * 1152 + head * 96;
  const float* k0 = base + 0 * 1152 + 384 + head * 96;
  const float* k1 = base + 1 * 1152 + 384 + head * 96;
  const float* k2 = base + 2 * 1152 + 384 + head * 96;
  float s0 = 0.f, s1 = 0.f, s2 = 0.f;
  for (int d = 0; d < 96; ++d) {
    float qd = q[d];
    s0 += qd * k0[d]; s1 += qd * k1[d]; s2 += qd * k2[d];
  }
  const float sc = 0.1020620726f;  // 1/sqrt(96)
  s0 *= sc; s1 *= sc; s2 *= sc;
  float mx = fmaxf(s0, fmaxf(s1, s2));
  float e0 = expf(s0 - mx), e1 = expf(s1 - mx), e2 = expf(s2 - mx);
  float inv = 1.f / (e0 + e1 + e2);
  float a0 = e0 * inv, a1 = e1 * inv, a2 = e2 * inv;
  const float* v0 = base + 0 * 1152 + 768 + head * 96;
  const float* v1 = base + 1 * 1152 + 768 + head * 96;
  const float* v2 = base + 2 * 1152 + 768 + head * 96;
  float* o = ao + (size_t)(n * 3 + ti) * 384 + head * 96;
  for (int d = 0; d < 96; ++d) o[d] = a0 * v0[d] + a1 * v1[d] + a2 * v2[d];
}

// per-batch mean of hcat (tokens viewed as (BN, 384, 3))
__global__ void k_batch_mean(const float* __restrict__ tok, float* __restrict__ mv,
                             int N) {
  int gid = blockIdx.x * blockDim.x + threadIdx.x;
  if (gid >= 2 * 1152) return;
  int b = gid / 1152, q = gid % 1152;
  float s = 0.f;
  const float* p = tok + (size_t)b * N * 1152 + q;
  for (int n = 0; n < N; ++n) s += p[(size_t)n * 1152];
  mv[gid] = s / (float)N;
}

__global__ void k_zeq_mean(const float* __restrict__ zn, float* __restrict__ out,
                           int N) {
  int gid = blockIdx.x * blockDim.x + threadIdx.x;
  if (gid >= 2 * 255) return;
  int b = gid / 255, q = gid % 255;
  float s = 0.f;
  const float* p = zn + (size_t)b * N * 255 + q;
  for (int n = 0; n < N; ++n) s += p[(size_t)n * 255];
  out[gid] = s / (float)N;
}

// inv features: per (b, c<768): mean over nodes of norms (c<384) or dots
__global__ void k_inv(const float* __restrict__ tok, const float* __restrict__ mv,
                      float* __restrict__ inv, int N) {
  int gid = blockIdx.x * blockDim.x + threadIdx.x;
  if (gid >= 2 * 768) return;
  int b = gid / 768, c = gid % 768;
  float s = 0.f;
  if (c < 384) {
    const float* p = tok + (size_t)b * N * 1152 + (size_t)c * 3;
    for (int n = 0; n < N; ++n) {
      const float* q = p + (size_t)n * 1152;
      s += sqrtf(q[0] * q[0] + q[1] * q[1] + q[2] * q[2]);
    }
  } else {
    int cc = c - 384;
    float m0 = mv[b * 1152 + cc * 3], m1 = mv[b * 1152 + cc * 3 + 1],
          m2 = mv[b * 1152 + cc * 3 + 2];
    const float* p = tok + (size_t)b * N * 1152 + (size_t)cc * 3;
    for (int n = 0; n < N; ++n) {
      const float* q = p + (size_t)n * 1152;
      s += q[0] * m0 + q[1] * m1 + q[2] * m2;
    }
  }
  inv[gid] = s / (float)N;
}

// final heads in one workgroup
__global__ void k_head(const float* __restrict__ inv, const float* __restrict__ W1,
                       const float* __restrict__ b1, const float* __restrict__ g,
                       const float* __restrict__ bb, const float* __restrict__ W2,
                       const float* __restrict__ b2, const float* __restrict__ aW1,
                       const float* __restrict__ ab1, const float* __restrict__ aW2,
                       const float* __restrict__ ab2, float* __restrict__ out) {
  __shared__ float sInv[1536];
  __shared__ float sY[512];
  __shared__ float sA[128];
  int t = threadIdx.x;
  for (int i = t; i < 1536; i += 256) sInv[i] = inv[i];
  __syncthreads();
  for (int b = 0; b < 2; ++b) {
    float acc = b1[t];
    const float* wr = W1 + (size_t)t * 768;
    const float* iv = sInv + b * 768;
    for (int k = 0; k < 768; ++k) acc += iv[k] * wr[k];
    sY[b * 256 + t] = acc;
  }
  __syncthreads();
  {
    float y0 = sY[t], y1 = sY[256 + t];
    float mu = 0.5f * (y0 + y1);
    float var = 0.5f * ((y0 - mu) * (y0 - mu) + (y1 - mu) * (y1 - mu));
    float is = rsqrtf(var + 1e-5f);
    float a0 = (y0 - mu) * is * g[t] + bb[t];
    float a1 = (y1 - mu) * is * g[t] + bb[t];
    sY[t]       = (a0 >= 0.f) ? a0 : 0.2f * a0;
    sY[256 + t] = (a1 >= 0.f) ? a1 : 0.2f * a1;
  }
  __syncthreads();
  for (int b = 0; b < 2; ++b) {
    float acc = b2[t];
    const float* wr = W2 + (size_t)t * 256;
    const float* yv = sY + b * 256;
    for (int k = 0; k < 256; ++k) acc += yv[k] * wr[k];
    out[b * 256 + t] = acc;
  }
  if (t < 128) {
    int b = t >> 6, j = t & 63;
    float acc = ab1[j];
    const float* wr = aW1 + (size_t)j * 768;
    const float* iv = sInv + b * 768;
    for (int k = 0; k < 768; ++k) acc += iv[k] * wr[k];
    sA[t] = (acc >= 0.f) ? acc : 0.2f * acc;
  }
  __syncthreads();
  if (t < 2) {
    float acc = ab2[0];
    for (int k = 0; k < 64; ++k) acc += sA[t * 64 + k] * aW2[k];
    out[1022 + t] = 1.f / (1.f + expf(-acc));
  }
}

// ---------------------------------------------------------------------------
static inline unsigned gdiv(long n) { return (unsigned)((n + TPB - 1) / TPB); }
static inline unsigned gtilesMT(long m, long n, int MT) {
  long mt = (m + 15) / 16, nt = (n + 15) / 16;
  long tiles = ((mt + MT - 1) / MT) * nt;
  return (unsigned)((tiles + 7) / 8);  // 8 waves / block
}

extern "C" void kernel_launch(void* const* d_in, const int* in_sizes, int n_in,
                              void* d_out, int out_size, void* d_ws, size_t ws_size,
                              hipStream_t stream) {
  const int BN = 4096, NB = 2048;
  const float* X = (const float*)d_in[0];
  const int* SRC[3] = {(const int*)d_in[1], (const int*)d_in[3], (const int*)d_in[5]};
  const int* TGT[3] = {(const int*)d_in[2], (const int*)d_in[4], (const int*)d_in[6]};
  int EA[3] = {in_sizes[1], in_sizes[3], in_sizes[5]};
  const int P = 7;
  auto LP = [&](int s, int l, int k) {
    return (const float*)d_in[P + (s * 3 + l) * 12 + k];
  };
  const int PA = P + 108;
  const float* attn_inW  = (const float*)d_in[PA + 0];
  const float* attn_inb  = (const float*)d_in[PA + 1];
  const float* attn_outW = (const float*)d_in[PA + 2];
  const float* attn_outb = (const float*)d_in[PA + 3];
  const float* eqproj_W  = (const float*)d_in[PA + 4];
  const float* invh_W1   = (const float*)d_in[PA + 5];
  const float* invh_b1   = (const float*)d_in[PA + 6];
  const float* bn_g      = (const float*)d_in[PA + 7];
  const float* bn_b      = (const float*)d_in[PA + 8];
  const float* invh_W2   = (const float*)d_in[PA + 9];
  const float* invh_b2   = (const float*)d_in[PA + 10];
  const float* an_W1     = (const float*)d_in[PA + 11];
  const float* an_b1     = (const float*)d_in[PA + 12];
  const float* an_W2     = (const float*)d_in[PA + 13];
  const float* an_b2     = (const float*)d_in[PA + 14];

  size_t maxE = (size_t)EA[0];
  if ((size_t)EA[1] > maxE) maxE = EA[1];
  if ((size_t)EA[2] > maxE) maxE = EA[2];

  // workspace layout (floats)
  float* w = (float*)d_ws;
  size_t o = 0;
  float* dist    = w + o; o += maxE;
  float* sumdist = w + o; o += BN;   // cnt contiguous after sumdist
  float* cnt     = w + o; o += BN;
  float* ldv     = w + o; o += BN;
  float* stat    = w + o; o += 256;
  float* hA      = w + o; o += (size_t)BN * 384;
  float* hB      = w + o; o += (size_t)BN * 384;
  float* tokens  = w + o; o += (size_t)BN * 1152;
  float* invbuf  = w + o; o += 1536;
  float* meanvec = w + o; o += 2304;
  size_t szC = (size_t)maxE * 128; if ((size_t)BN * 255 > szC) szC = (size_t)BN * 255;
  float* regC = w + o; o += szC;           // mod  / zeq_nodes
  size_t szA = (size_t)maxE * 384; if ((size_t)BN * 3456 > szA) szA = (size_t)BN * 3456;
  float* regA = w + o; o += szA;           // x1   / qkv
  size_t szB = (size_t)maxE * 384; if ((size_t)BN * 1152 > szB) szB = (size_t)BN * 1152;
  float* regB = w + o; o += szB;           // x2   / attn_o
  o = (o + 3) & ~(size_t)3;                // align f16 arena to 16B
  _Float16* warena = (_Float16*)(w + o);
  (void)ws_size; (void)n_in; (void)out_size;

  // f16 weight conversion (padded M->128, K->32)
  size_t ho = 0;
  auto conv = [&](const float* W, int M, int K) -> const _Float16* {
    int Mp = (M + 127) & ~127;
    int Kp = (K + 31) & ~31;
    _Float16* dst = warena + ho;
    ho += (size_t)Mp * Kp;
    long n = (long)Mp * Kp;
    k_w2h<<<gdiv(n), TPB, 0, stream>>>(W, dst, M, K, Kp, n);
    return dst;
  };
  const _Float16 *hW_e1[3][3], *hW_e2[3][3];
  for (int s = 0; s < 3; ++s) {
    int in_ch = 1;
    for (int l = 0; l < 3; ++l) {
      int C = 32 << l;
      hW_e1[s][l] = conv(LP(s, l, 6), C, 2 * in_ch + 1);
      hW_e2[s][l] = conv(LP(s, l, 9), C, C);
      in_ch = C;
    }
  }
  const _Float16* hW_ain  = conv(attn_inW, 1152, 384);
  const _Float16* hW_aout = conv(attn_outW, 384, 384);
  const _Float16* hW_eq   = conv(eqproj_W, 85, 384);

  float* out = (float*)d_out;

  // ---- multi-scale edge-conv stack -------------------------------------
  for (int s = 0; s < 3; ++s) {
    const int* src = SRC[s];
    const int* tgt = TGT[s];
    int E = EA[s];
    long N3 = 3L * E;
    k_zero<<<gdiv(2 * BN), TPB, 0, stream>>>(sumdist, 2 * BN);
    k_edge_geom<<<gdiv(E), TPB, 0, stream>>>(src, tgt, X, dist, sumdist, cnt, E);
    k_node_ld<<<gdiv(BN), TPB, 0, stream>>>(sumdist, cnt, ldv, BN);

    const float* hcur = X;  // pos == (BN,3,1) layout
    int in_ch = 1;
    for (int l = 0; l < 3; ++l) {
      int C = 32 << l;
      int mlp_in = 2 * in_ch + 1;
      int Kp = (mlp_in + 31) & ~31;
      float* mod = regC; float* x1 = regA; float* x2 = regB;

      k_mod<<<gdiv(E), TPB, 0, stream>>>(src, dist, ldv, LP(s, l, 0), LP(s, l, 1),
                                         LP(s, l, 2), LP(s, l, 3), LP(s, l, 4),
                                         LP(s, l, 5), mod, E, C);
      // vn block 1 (fused gather GEMM) -> x1 (3E, C); MT = C/16
      if (C == 32)
        k_vn1<2><<<gtilesMT(C, N3, 2), TPB, 0, stream>>>(hW_e1[s][l], hcur, src, tgt,
                                                         X, dist, x1, C, (int)N3, Kp,
                                                         in_ch);
      else if (C == 64)
        k_vn1<4><<<gtilesMT(C, N3, 4), TPB, 0, stream>>>(hW_e1[s][l], hcur, src, tgt,
                                                         X, dist, x1, C, (int)N3, Kp,
                                                         in_ch);
      else
        k_vn1<8><<<gtilesMT(C, N3, 8), TPB, 0, stream>>>(hW_e1[s][l], hcur, src, tgt,
                                                         X, dist, x1, C, (int)N3, Kp,
                                                         in_ch);
      k_zero<<<1, TPB, 0, stream>>>(stat, 256);
      k_vn_stats<<<gdiv((long)E * C), TPB, 0, stream>>>(x1, stat, E, C);
      k_vn_norm<<<gdiv((long)E * C), TPB, 0, stream>>>(x1, stat, LP(s, l, 7),
                                                       LP(s, l, 8), E, C);
      // vn block 2: x2 (3E, C) = x1 (3E, C) @ W^T; MT = C/16
      if (C == 32)
        k_gemm<0, 0, 2, false, false><<<gtilesMT(C, N3, 2), TPB, 0, stream>>>(
            hW_e2[s][l], x1, x2, nullptr, C, (int)N3, C, C, C, 0);
      else if (C == 64)
        k_gemm<0, 0, 4, false, false><<<gtilesMT(C, N3, 4), TPB, 0, stream>>>(
            hW_e2[s][l], x1, x2, nullptr, C, (int)N3, C, C, C, 0);
      else
        k_gemm<0, 0, 8, false, false><<<gtilesMT(C, N3, 8), TPB, 0, stream>>>(
            hW_e2[s][l], x1, x2, nullptr, C, (int)N3, C, C, C, 0);
      k_zero<<<1, TPB, 0, stream>>>(stat, 256);
      k_vn_stats<<<gdiv((long)E * C), TPB, 0, stream>>>(x2, stat, E, C);
      k_vn_norm<<<gdiv((long)E * C), TPB, 0, stream>>>(x2, stat, LP(s, l, 10),
                                                       LP(s, l, 11), E, C);
      // modulate + segment mean back to nodes
      float* hn = (l & 1) ? hB : hA;
      k_zero<<<gdiv((long)BN * C * 3), TPB, 0, stream>>>(hn, (long)BN * C * 3);
      k_scatter<<<gdiv((long)E * C), TPB, 0, stream>>>(x2, mod, src, hn, E, C);
      k_divcnt<<<gdiv((long)BN * C), TPB, 0, stream>>>(hn, cnt, BN, C);
      hcur = hn;
      in_ch = C;
    }
    k_copy_tok<<<gdiv((long)BN * 384), TPB, 0, stream>>>(hcur, tokens, s, BN);
  }

  // ---- multi-scale token attention (T=3, E=384) ------------------------
  float* qkv    = regA;
  float* attn_o = regB;
  k_gemm<0, 0, 8, true, false><<<gtilesMT(1152, 12288, 8), TPB, 0, stream>>>(
      hW_ain, tokens, qkv, attn_inb, 1152, 12288, 384, 384, 1152, 0);
  k_attn<<<gdiv(BN * 12), TPB, 0, stream>>>(qkv, attn_o, BN);
  k_gemm<0, 0, 8, true, true><<<gtilesMT(384, 12288, 8), TPB, 0, stream>>>(
      hW_aout, attn_o, tokens, attn_outb, 384, 12288, 384, 384, 384, 0);

  // ---- equivariant projection + batch reductions -----------------------
  float* zeq = regC;
  k_gemm<1, 1, 2, false, false><<<gtilesMT(85, 12288, 2), TPB, 0, stream>>>(
      hW_eq, tokens, zeq, nullptr, 85, 12288, 384, 0, 0, 85);
  k_zeq_mean<<<gdiv(510), TPB, 0, stream>>>(zeq, out + 512, NB);
  k_batch_mean<<<gdiv(2304), TPB, 0, stream>>>(tokens, meanvec, NB);
  k_inv<<<gdiv(1536), TPB, 0, stream>>>(tokens, meanvec, invbuf, NB);

  // ---- invariant heads (single workgroup) ------------------------------
  k_head<<<1, TPB, 0, stream>>>(invbuf, invh_W1, invh_b1, bn_g, bn_b, invh_W2,
                                invh_b2, an_W1, an_b1, an_W2, an_b2, out);
}
